// ExpertModuleTrm_7267084665600
// MI455X (gfx1250) — matrix operand order repaired and checked by reading, not compiled, not run on hardware
//
#include <hip/hip_runtime.h>
#include <hip/hip_bf16.h>
#include <stdint.h>

// Problem constants (from reference)
#define B_N   16
#define L_N   2048
#define D_N   400
#define H_N   100     // real head size
#define HP_N  128     // padded head size (zero-padded; exact math)
#define DP_N  416     // padded model dim (13 chunks of 32)
#define F_N   400
#define FP_N  512     // padded h-feature dim for wfT
#define U0_N  512
#define U1_N  256
#define SCALE_F 0.1f  // 1/sqrt(100)

// Padded LDS strides (halfwords), all rows 16B-aligned for b128 async copies
// and odd-ish dword strides for conflict-free fragment reads.
#define RS_T  136     // full-K (128) tiles: 272B rows
#define RS_C  40      // 32-k chunk tiles:  80B rows
#define RS_P  34      // P scratch (ds-written, no async)

typedef __bf16  bf16_t;
typedef bf16_t  v16bf __attribute__((ext_vector_type(16)));
typedef float   v8f   __attribute__((ext_vector_type(8)));
typedef int     v4i   __attribute__((vector_size(16)));   // matches builtin proto

union FragU { unsigned u[8]; v16bf v; };

// ---------------------------------------------------------------------------
// CDNA5 async global->LDS copies (ASYNCcnt) with synchronous fallback
// ---------------------------------------------------------------------------
#if __has_builtin(__builtin_amdgcn_global_load_async_to_lds_b128) && \
    __has_builtin(__builtin_amdgcn_s_wait_asynccnt)
#define HAVE_ASYNC_LDS 1
#endif

__device__ __forceinline__ void copy16_async(const void* g, void* l) {
#ifdef HAVE_ASYNC_LDS
  __builtin_amdgcn_global_load_async_to_lds_b128(
      (__attribute__((address_space(1))) v4i*)g,
      (__attribute__((address_space(3))) v4i*)l, 0, 0);
#else
  *(uint4*)l = *(const uint4*)g;
#endif
}

__device__ __forceinline__ void async_wait0() {
#ifdef HAVE_ASYNC_LDS
  __builtin_amdgcn_s_wait_asynccnt(0);
#endif
}

// Stage a 32-row x 256B tile (bf16 [32][128]) into LDS with RS_T stride.
__device__ __forceinline__ void stage_tile_async(const unsigned short* __restrict__ gsrc,
                                                 unsigned short* __restrict__ ldst,
                                                 int tid) {
#pragma unroll
  for (int i = 0; i < 4; ++i) {
    int idx = tid + i * 128;           // 512 chunks of 16B
    int r = idx >> 4, c = idx & 15;
    copy16_async(gsrc + (size_t)r * HP_N + c * 8, ldst + r * RS_T + c * 8);
  }
}

// ---------------------------------------------------------------------------
__device__ __forceinline__ unsigned short f2bf(float x) {
  unsigned u = __float_as_uint(x);
  u += 0x7FFFu + ((u >> 16) & 1u);            // round-to-nearest-even
  return (unsigned short)(u >> 16);
}

__device__ __forceinline__ unsigned f2bf_pk(float lo, float hi) {
  return (unsigned)f2bf(lo) | ((unsigned)f2bf(hi) << 16);
}

__device__ __forceinline__ v8f vzero8() {
  v8f z = {0.f,0.f,0.f,0.f,0.f,0.f,0.f,0.f};
  return z;
}

// A fragment (bf16 16x32): row = lane&15; VGPR i holds packed K pair
// k0 = (lane>=16?16:0) + 2i.  base = row-major tile, rs halfwords/row.
__device__ __forceinline__ v16bf load_a_frag(const unsigned short* base, int rs) {
  int lane = threadIdx.x & 31;
  int row  = lane & 15;
  int kh   = (lane & 16) ? 16 : 0;
  FragU f;
#pragma unroll
  for (int i = 0; i < 8; ++i) {
    int k0 = kh + i * 2;
    f.u[i] = *(const unsigned*)(base + row * rs + k0);
  }
  return f.v;
}

// B fragment (bf16 32x16): col = lane&15; VGPR i holds K pair k0 = (lane>=16?16:0)+2i.
// baseT laid out [col][k] (k contiguous), stride cs halfwords.
__device__ __forceinline__ v16bf load_b_frag(const unsigned short* baseT, int cs) {
  int lane = threadIdx.x & 31;
  int col  = lane & 15;
  int kh   = (lane & 16) ? 16 : 0;
  FragU f;
#pragma unroll
  for (int i = 0; i < 8; ++i) {
    f.u[i] = *(const unsigned*)(baseT + col * cs + kh + i * 2);
  }
  return f.v;
}

// B fragment from natural [k][n] layout (k-pairs strided): two 16-bit loads/dword.
__device__ __forceinline__ v16bf load_b_frag_strided(const unsigned short* baseN, int rs) {
  int lane = threadIdx.x & 31;
  int n    = lane & 15;
  int kh   = (lane & 16) ? 16 : 0;
  FragU f;
#pragma unroll
  for (int i = 0; i < 8; ++i) {
    unsigned lo = baseN[(kh + 2 * i) * rs + n];
    unsigned hi = baseN[(kh + 2 * i + 1) * rs + n];
    f.u[i] = lo | (hi << 16);
  }
  return f.v;
}

__device__ __forceinline__ v8f wmma_bf16(v16bf a, v16bf b, v8f c) {
  return __builtin_amdgcn_wmma_f32_16x16x32_bf16(false, a, false, b, (short)0, c,
                                                 false, false);
}

__device__ __forceinline__ float redmax16(float x) {
#pragma unroll
  for (int d = 8; d >= 1; d >>= 1) x = fmaxf(x, __shfl_xor(x, d, 32));
  return x;
}
__device__ __forceinline__ float redsum16(float x) {
#pragma unroll
  for (int d = 8; d >= 1; d >>= 1) x += __shfl_xor(x, d, 32);
  return x;
}

// ---------------------------------------------------------------------------
// One-shot conversion pre-passes (hoist all f32->bf16 out of GEMM hot loops)
// ---------------------------------------------------------------------------
// x f32 [32768][400] -> xbf bf16 [32768][416] zero-padded (read x from HBM once)
__global__ void xcvt_kernel(const float* __restrict__ x, unsigned* __restrict__ xbf) {
  int i = blockIdx.x * 256 + threadIdx.x;            // one u32 (2 elems) per thread
  if (i >= (B_N * L_N) * (DP_N / 2)) return;
  int row = i / (DP_N / 2), cp = i % (DP_N / 2), c = cp * 2;
  float v0 = (c     < D_N) ? x[(size_t)row * D_N + c]     : 0.f;
  float v1 = (c + 1 < D_N) ? x[(size_t)row * D_N + c + 1] : 0.f;
  xbf[(size_t)row * (DP_N / 2) + cp] = f2bf_pk(v0, v1);
}

// kern f32 [3][400][100] -> wqkvT bf16 [3][128 cols][416 k] (B-fragment-ready)
__global__ void wqkv_cvt_kernel(const float* __restrict__ kern, unsigned* __restrict__ wT) {
  int i = blockIdx.x * 256 + threadIdx.x;
  if (i >= 3 * HP_N * (DP_N / 2)) return;
  int w = i / (HP_N * (DP_N / 2));
  int r = i % (HP_N * (DP_N / 2));
  int n = r / (DP_N / 2), cp = r % (DP_N / 2), k = cp * 2;
  float v0 = (n < H_N && k     < D_N) ? kern[(size_t)w * D_N * H_N + (size_t)k * H_N + n]       : 0.f;
  float v1 = (n < H_N && k + 1 < D_N) ? kern[(size_t)w * D_N * H_N + (size_t)(k + 1) * H_N + n] : 0.f;
  wT[i] = f2bf_pk(v0, v1);
}

// wf f32 [128][400] -> wfT bf16 [512 cols][128 k] (B-fragment-ready, zero-padded)
__global__ void wf_cvt_kernel(const float* __restrict__ wf, unsigned* __restrict__ wfT) {
  int i = blockIdx.x * 256 + threadIdx.x;
  if (i >= FP_N * (HP_N / 2)) return;
  int n = i / (HP_N / 2), cp = i % (HP_N / 2), k = cp * 2;
  float v0 = (n < F_N) ? wf[(size_t)k * F_N + n]       : 0.f;
  float v1 = (n < F_N) ? wf[(size_t)(k + 1) * F_N + n] : 0.f;
  wfT[i] = f2bf_pk(v0, v1);
}

// ---------------------------------------------------------------------------
// Kernel: QKV projection.  grid (512, 3): x[64-row tile] x {q,k,v}.
// Double-buffered async staging of pre-converted bf16 x / transposed weights.
// Output bf16 [B*L, 128]; cols 100..127 zero (weight tile zero-padded).
// ---------------------------------------------------------------------------
__global__ __launch_bounds__(128) void qkv_kernel(const unsigned short* __restrict__ xbf,
                                                  const unsigned short* __restrict__ wqkvT,
                                                  unsigned short* __restrict__ qkv) {
  __shared__ __attribute__((aligned(16))) unsigned short x_lds[2][64 * RS_C];
  __shared__ __attribute__((aligned(16))) unsigned short w_lds[2][128 * RS_C];

  const int tid = threadIdx.x, lane = tid & 31, wv = tid >> 5;
  const int m = blockIdx.x;                     // 64-row tile of flattened [32768]
  const unsigned short* xg = xbf + (size_t)m * 64 * DP_N;
  const unsigned short* wg = wqkvT + (size_t)blockIdx.y * HP_N * DP_N;
  unsigned short* out = qkv + (size_t)blockIdx.y * B_N * L_N * HP_N;

  v8f acc[8];
#pragma unroll
  for (int i = 0; i < 8; ++i) acc[i] = vzero8();

  // async-stage one 32-k chunk (x: 64x64B, w: 128x64B)
  auto stage_chunk = [&](int kc, int buf) {
#pragma unroll
    for (int i = 0; i < 2; ++i) {               // x tile: 256 chunks of 16B
      int idx = tid + i * 128;
      int r = idx >> 2, c = idx & 3;
      copy16_async(xg + (size_t)r * DP_N + kc * 32 + c * 8, x_lds[buf] + r * RS_C + c * 8);
    }
#pragma unroll
    for (int i = 0; i < 4; ++i) {               // w tile: 512 chunks of 16B
      int idx = tid + i * 128;
      int n = idx >> 2, c = idx & 3;
      copy16_async(wg + (size_t)n * DP_N + kc * 32 + c * 8, w_lds[buf] + n * RS_C + c * 8);
    }
  };

  stage_chunk(0, 0);
  for (int kc = 0; kc < 13; ++kc) {             // K = 416 -> 13 chunks of 32
    async_wait0();
    __syncthreads();
    if (kc + 1 < 13) stage_chunk(kc + 1, (kc + 1) & 1);
    v16bf a = load_a_frag(x_lds[kc & 1] + wv * 16 * RS_C, RS_C);
#pragma unroll
    for (int nt = 0; nt < 8; ++nt)
      acc[nt] = wmma_bf16(a, load_b_frag(w_lds[kc & 1] + nt * 16 * RS_C, RS_C), acc[nt]);
  }

  const int hi8 = (lane & 16) >> 1, col = lane & 15;
#pragma unroll
  for (int nt = 0; nt < 8; ++nt)
#pragma unroll
    for (int i2 = 0; i2 < 8; ++i2) {
      int row = m * 64 + wv * 16 + i2 + hi8;
      out[(size_t)row * HP_N + nt * 16 + col] = f2bf(acc[nt][i2]);
    }
}

// ---------------------------------------------------------------------------
// Kernel: flash attention.  grid (32, 16) = (qtile, batch), 4 waves x 16 queries.
// Double-buffered async K/V staging; online softmax; head -> bf16 [B*L,128].
// ---------------------------------------------------------------------------
__global__ __launch_bounds__(128) void attn_kernel(const unsigned short* __restrict__ q,
                                                   const unsigned short* __restrict__ k,
                                                   const unsigned short* __restrict__ v,
                                                   unsigned short* __restrict__ head) {
  __shared__ __attribute__((aligned(16))) unsigned short k_lds[2][32 * RS_T]; // [key][hd]
  __shared__ __attribute__((aligned(16))) unsigned short v_lds[2][32 * RS_T]; // [key][hd]
  __shared__ __attribute__((aligned(16))) unsigned short p_lds[4 * 16 * RS_P];

  const int tid = threadIdx.x, lane = tid & 31, wv = tid >> 5;
  const int b = blockIdx.y, qt = blockIdx.x;
  const unsigned short* qg = q + (size_t)b * L_N * HP_N;
  const unsigned short* kg = k + (size_t)b * L_N * HP_N;
  const unsigned short* vg = v + (size_t)b * L_N * HP_N;
  unsigned short* hg = head + (size_t)b * L_N * HP_N;

  // Resident Q fragments: 16 queries x 128 head-dim per wave
  v16bf qa[4];
  {
    const unsigned short* qrow = qg + (size_t)(qt * 64 + wv * 16) * HP_N;
#pragma unroll
    for (int hc = 0; hc < 4; ++hc) qa[hc] = load_a_frag(qrow + hc * 32, HP_N);
  }

  v8f O[8];
  float m_[8], l_[8];
#pragma unroll
  for (int i = 0; i < 8; ++i) { O[i] = vzero8(); m_[i] = -1e30f; l_[i] = 0.f; }

  // Prologue: DMA tile 0
  stage_tile_async(kg, k_lds[0], tid);
  stage_tile_async(vg, v_lds[0], tid);

  for (int kt = 0; kt < 64; ++kt) {             // 64 tiles x 32 keys
    async_wait0();                               // own tile-kt copies landed
    __syncthreads();                             // everyone's copies visible

    if (kt + 1 < 64) {                           // DMA next tile into other buffer
      stage_tile_async(kg + (size_t)(kt + 1) * 32 * HP_N, k_lds[(kt + 1) & 1], tid);
      stage_tile_async(vg + (size_t)(kt + 1) * 32 * HP_N, v_lds[(kt + 1) & 1], tid);
    }
    const unsigned short* kl = k_lds[kt & 1];
    const unsigned short* vl = v_lds[kt & 1];

    // S = Q K^T  (two 16-key sub-tiles, K reduction over 128 head dims)
    v8f S0 = vzero8(), S1 = vzero8();
#pragma unroll
    for (int hc = 0; hc < 4; ++hc) {
      S0 = wmma_bf16(qa[hc], load_b_frag(kl + hc * 32, RS_T), S0);
      S1 = wmma_bf16(qa[hc], load_b_frag(kl + 16 * RS_T + hc * 32, RS_T), S1);
    }

    // Online softmax; write P (bf16) into wave-private LDS in A-layout source form
    float alpha[8];
    unsigned short* pl = p_lds + wv * 16 * RS_P;
    const int hi8 = (lane & 16) >> 1, col = lane & 15;
#pragma unroll
    for (int i2 = 0; i2 < 8; ++i2) {
      float s0 = S0[i2] * SCALE_F, s1 = S1[i2] * SCALE_F;
      float mx = redmax16(fmaxf(s0, s1));
      float nm = fmaxf(m_[i2], mx);
      float al = __expf(m_[i2] - nm);
      float p0 = __expf(s0 - nm), p1 = __expf(s1 - nm);
      float rs = redsum16(p0 + p1);
      l_[i2] = l_[i2] * al + rs;
      m_[i2] = nm;
      alpha[i2] = al;
      int row = i2 + hi8;
      pl[row * RS_P + col]      = f2bf(p0);
      pl[row * RS_P + col + 16] = f2bf(p1);
    }
#pragma unroll
    for (int nt = 0; nt < 8; ++nt)
#pragma unroll
      for (int i2 = 0; i2 < 8; ++i2) O[nt][i2] = O[nt][i2] * alpha[i2];

    // O += P V   (wave-internal LDS ops are in-order; no barrier needed for pl)
    v16bf pa = load_a_frag(pl, RS_P);
#pragma unroll
    for (int nt = 0; nt < 8; ++nt)
      O[nt] = wmma_bf16(pa, load_b_frag_strided(vl + nt * 16, RS_T), O[nt]);
  }

  const int hi8 = (lane & 16) >> 1, col = lane & 15;
#pragma unroll
  for (int i2 = 0; i2 < 8; ++i2) {
    float inv = 1.0f / l_[i2];
    int row = qt * 64 + wv * 16 + i2 + hi8;
#pragma unroll
    for (int nt = 0; nt < 8; ++nt)
      hg[(size_t)row * HP_N + nt * 16 + col] = f2bf(O[nt][i2] * inv);
  }
}

// ---------------------------------------------------------------------------
// Kernel: h = relu(head @ Wfused + b2) with fused max/sum pooling.
// grid (512, 4): 64-row tile x 128-col tile (last tile has 16 valid cols).
// Head tile and full weight tile are async-DMA'd once; no barriers in K loop.
// ---------------------------------------------------------------------------
__global__ __launch_bounds__(128) void hpool_kernel(const unsigned short* __restrict__ head,
                                                    const unsigned short* __restrict__ wfT,
                                                    const float* __restrict__ b2,
                                                    unsigned* __restrict__ hmax,
                                                    float* __restrict__ psum) {
  __shared__ __attribute__((aligned(16))) unsigned short h_lds[64 * RS_T];
  __shared__ __attribute__((aligned(16))) unsigned short wt_lds[128 * RS_T];
  __shared__ float rmax_lds[4 * 128];
  __shared__ float rsum_lds[4 * 128];

  const int tid = threadIdx.x, lane = tid & 31, wv = tid >> 5;
  const int m = blockIdx.x, nt4 = blockIdx.y;
  const int b = m >> 5;                          // 32 row-tiles per batch
  const int nbase = nt4 * 128;
  const int ncols = (F_N - nbase < 128) ? (F_N - nbase) : 128;
  const int nsub = ncols >> 4;

  // DMA the full 64x128 bf16 head tile and the 128-col x 128-k weight tile
  const unsigned short* hsrc = head + (size_t)m * 64 * HP_N;
#pragma unroll
  for (int i = 0; i < 8; ++i) {
    int idx = tid + i * 128;                     // 1024 chunks of 16B
    int r = idx >> 4, c = idx & 15;
    copy16_async(hsrc + (size_t)r * HP_N + c * 8, h_lds + r * RS_T + c * 8);
  }
  const unsigned short* wsrc = wfT + (size_t)nbase * HP_N;
#pragma unroll
  for (int i = 0; i < 16; ++i) {
    int idx = tid + i * 128;                     // 2048 chunks of 16B
    int r = idx >> 4, c = idx & 15;
    copy16_async(wsrc + (size_t)r * HP_N + c * 8, wt_lds + r * RS_T + c * 8);
  }
  async_wait0();
  __syncthreads();

  v8f acc[8];
#pragma unroll
  for (int i = 0; i < 8; ++i) acc[i] = vzero8();

#pragma unroll
  for (int hc = 0; hc < 4; ++hc) {               // K = 128 -> 4 chunks
    v16bf a = load_a_frag(h_lds + wv * 16 * RS_T + hc * 32, RS_T);
    for (int nt = 0; nt < nsub; ++nt)
      acc[nt] = wmma_bf16(a, load_b_frag(wt_lds + nt * 16 * RS_T + hc * 32, RS_T), acc[nt]);
  }

  // epilogue: bias + relu + per-block max/sum reduction (h never hits memory)
  const int col = lane & 15;
  for (int nt = 0; nt < nsub; ++nt) {
    float bias = b2[nbase + nt * 16 + col];
    float tmax = 0.f, tsum = 0.f;
#pragma unroll
    for (int i2 = 0; i2 < 8; ++i2) {
      float h = fmaxf(acc[nt][i2] + bias, 0.f);
      tmax = fmaxf(tmax, h);
      tsum += h;
    }
    tmax = fmaxf(tmax, __shfl_xor(tmax, 16, 32));
    tsum += __shfl_xor(tsum, 16, 32);
    if (lane < 16) {
      rmax_lds[wv * 128 + nt * 16 + col] = tmax;
      rsum_lds[wv * 128 + nt * 16 + col] = tsum;
    }
  }
  __syncthreads();
  if (tid < 128 && tid < ncols) {
    int cn = tid;
    float mx = rmax_lds[cn], sm = rsum_lds[cn];
#pragma unroll
    for (int w2i = 1; w2i < 4; ++w2i) {
      mx = fmaxf(mx, rmax_lds[w2i * 128 + cn]);
      sm += rsum_lds[w2i * 128 + cn];
    }
    atomicMax(&hmax[b * F_N + nbase + cn], __float_as_uint(mx));  // h >= 0: bits monotonic
    psum[(size_t)(b * 32 + (m & 31)) * F_N + nbase + cn] = sm;    // deterministic sums
  }
}

// ---------------------------------------------------------------------------
// Small kernels
// ---------------------------------------------------------------------------
__global__ void zero_hmax_kernel(unsigned* hmax) {
  int i = blockIdx.x * 256 + threadIdx.x;
  if (i < B_N * F_N) hmax[i] = 0u;
}

// Wfused[128][400] = (sum of 4 row-blocks of dense) @ w2; rows 100..127 zero.
__global__ void wf_kernel(const float* __restrict__ dense, const float* __restrict__ w2,
                          float* __restrict__ wf) {
  int i = blockIdx.x * 256 + threadIdx.x;
  if (i >= HP_N * F_N) return;
  int h = i / F_N, n = i % F_N;
  float acc = 0.f;
  if (h < H_N) {
    for (int o = 0; o < F_N; ++o) {
      float e = dense[(size_t)h * F_N + o] + dense[(size_t)(h + 100) * F_N + o] +
                dense[(size_t)(h + 200) * F_N + o] + dense[(size_t)(h + 300) * F_N + o];
      acc += e * w2[(size_t)o * F_N + n];
    }
  }
  wf[i] = acc;
}

__global__ void havg_kernel(const float* __restrict__ psum, float* __restrict__ havg) {
  int i = blockIdx.x * 256 + threadIdx.x;
  if (i >= B_N * F_N) return;
  int b = i / F_N, c = i % F_N;
  float s = 0.f;
#pragma unroll 4
  for (int j = 0; j < 32; ++j) s += psum[(size_t)(b * 32 + j) * F_N + c];
  havg[i] = s * (1.0f / 2048.0f);
}

__global__ void z1_kernel(const unsigned* __restrict__ hmax, const float* __restrict__ havg,
                          const float* __restrict__ w3, const float* __restrict__ b3,
                          float* __restrict__ z1) {
  int i = blockIdx.x * 256 + threadIdx.x;
  if (i >= B_N * U0_N) return;
  int b = i >> 9, u = i & (U0_N - 1);
  float acc = b3[u];
  for (int j = 0; j < F_N; ++j)
    acc += __uint_as_float(hmax[b * F_N + j]) * w3[(size_t)j * U0_N + u];
  for (int j = 0; j < F_N; ++j)
    acc += havg[b * F_N + j] * w3[(size_t)(F_N + j) * U0_N + u];
  z1[i] = fmaxf(acc, 0.f);
}

__global__ void z2_kernel(const float* __restrict__ z1, const float* __restrict__ w4,
                          const float* __restrict__ b4, float* __restrict__ out) {
  int i = blockIdx.x * 256 + threadIdx.x;
  if (i >= B_N * U1_N) return;
  int b = i >> 8, u = i & (U1_N - 1);
  float acc = b4[u];
  for (int j = 0; j < U0_N; ++j) acc += z1[b * U0_N + j] * w4[(size_t)j * U1_N + u];
  out[i] = fmaxf(acc, 0.f);
}

// ---------------------------------------------------------------------------
extern "C" void kernel_launch(void* const* d_in, const int* in_sizes, int n_in,
                              void* d_out, int out_size, void* d_ws, size_t ws_size,
                              hipStream_t stream) {
  (void)in_sizes; (void)n_in; (void)out_size; (void)ws_size;
  const float* x     = (const float*)d_in[0];
  const float* kern  = (const float*)d_in[1];
  const float* dense = (const float*)d_in[2];
  const float* w2    = (const float*)d_in[3];
  const float* b2    = (const float*)d_in[4];
  const float* w3    = (const float*)d_in[5];
  const float* b3    = (const float*)d_in[6];
  const float* w4    = (const float*)d_in[7];
  const float* b4    = (const float*)d_in[8];
  float* out = (float*)d_out;

  char* ws = (char*)d_ws;
  const size_t QKVB = (size_t)B_N * L_N * HP_N * 2;          // 8 MB per tensor
  unsigned short* q     = (unsigned short*)(ws);
  unsigned short* kk    = (unsigned short*)(ws + QKVB);
  unsigned short* v     = (unsigned short*)(ws + 2 * QKVB);
  unsigned short* head  = (unsigned short*)(ws + 3 * QKVB);
  unsigned short* xbf   = (unsigned short*)(ws + 4 * QKVB);  // 32768*416 bf16
  unsigned short* wqkvT = (unsigned short*)((char*)xbf + (size_t)B_N * L_N * DP_N * 2);
  float*    wf    = (float*)((char*)wqkvT + (size_t)3 * HP_N * DP_N * 2);
  unsigned short* wfT = (unsigned short*)((char*)wf + (size_t)HP_N * F_N * 4);
  unsigned* hmax  = (unsigned*)((char*)wfT + (size_t)FP_N * HP_N * 2);
  float*    psum  = (float*)((char*)hmax + (size_t)B_N * F_N * 4);
  float*    havg  = (float*)((char*)psum + (size_t)B_N * 32 * F_N * 4);
  float*    z1    = (float*)((char*)havg + (size_t)B_N * F_N * 4);

  zero_hmax_kernel<<<(B_N * F_N + 255) / 256, 256, 0, stream>>>(hmax);
  wf_kernel<<<(HP_N * F_N + 255) / 256, 256, 0, stream>>>(dense, w2, wf);
  wf_cvt_kernel<<<(FP_N * (HP_N / 2) + 255) / 256, 256, 0, stream>>>(wf, (unsigned*)wfT);
  xcvt_kernel<<<((B_N * L_N) * (DP_N / 2) + 255) / 256, 256, 0, stream>>>(x, (unsigned*)xbf);
  wqkv_cvt_kernel<<<(3 * HP_N * (DP_N / 2) + 255) / 256, 256, 0, stream>>>(kern, (unsigned*)wqkvT);
  qkv_kernel<<<dim3((B_N * L_N) / 64, 3), 128, 0, stream>>>(xbf, wqkvT, q);
  attn_kernel<<<dim3(L_N / 64, B_N), 128, 0, stream>>>(q, kk, v, head);
  hpool_kernel<<<dim3((B_N * L_N) / 64, 4), 128, 0, stream>>>(head, wfT, b2, hmax, psum);
  havg_kernel<<<(B_N * F_N + 255) / 256, 256, 0, stream>>>(psum, havg);
  z1_kernel<<<(B_N * U0_N + 255) / 256, 256, 0, stream>>>(hmax, havg, w3, b3, z1);
  z2_kernel<<<(B_N * U1_N + 255) / 256, 256, 0, stream>>>(z1, w4, b4, out);
}
